// CLSADecoder_15410342658409
// MI455X (gfx1250) — matrix operand order repaired and checked by reading, not compiled, not run on hardware
//
#include <hip/hip_runtime.h>
#include <hip/hip_bf16.h>
#include <math.h>

#define B_    128
#define T_    64
#define S_    128
#define D_    2048
#define CH_   32
#define COLS_ 8
#define K2D   4096   // 2*D

typedef __attribute__((ext_vector_type(16))) __bf16 v16bf;
typedef __attribute__((ext_vector_type(8)))  __bf16 v8bf;
typedef __attribute__((ext_vector_type(8)))  float  v8f;

__device__ __forceinline__ float sigm(float x) { return 1.0f / (1.0f + __expf(-x)); }

__device__ __forceinline__ float wave_sum(float v) {
#pragma unroll
  for (int off = 16; off > 0; off >>= 1) v += __shfl_down(v, off, 32);
  return v;
}

// ---------------- setup kernels (once per call) ----------------

__global__ void cvt_bf16_kernel(const float* __restrict__ src, __bf16* __restrict__ dst, int n) {
  int i = blockIdx.x * 256 + threadIdx.x;
  if (i < n) dst[i] = (__bf16)src[i];
}

// w: (K2D x D_) row-major  ->  wt: (D_ x K2D) row-major bf16 (N-major for WMMA B loads)
__global__ void cvt_transpose_kernel(const float* __restrict__ w, __bf16* __restrict__ wt) {
  int i = blockIdx.x * 256 + threadIdx.x;      // over D_*K2D
  int nn = i >> 12, k = i & (K2D - 1);
  wt[i] = (__bf16)w[(size_t)k * D_ + nn];
}

__global__ void init_state_kernel(const float* __restrict__ h0, const float* __restrict__ c0,
                                  float* __restrict__ hA, float* __restrict__ cA,
                                  float* __restrict__ hB, float* __restrict__ cB) {
  int i = blockIdx.x * 256 + threadIdx.x;      // B_*D_
  hA[i] = h0[i];           hB[i] = h0[(size_t)B_ * D_ + i];
  cA[i] = c0[i];           cB[i] = c0[(size_t)B_ * D_ + i];
}

// ---------------- ConvLSTM cell ----------------
// thread = (b, r, ch, col); computes 4 gates and the LSTM update. c updated in place,
// h written to hout (double-buffered). layer0: x from x_flat at step t; else x = (B,D) frame.
__global__ void convlstm_kernel(const float* __restrict__ x, int layer0, int t,
                                const float* __restrict__ hin,
                                float* __restrict__ c, float* __restrict__ hout,
                                const float* __restrict__ cxw, const float* __restrict__ cxb,
                                const float* __restrict__ chw, const float* __restrict__ chb) {
  int idx = blockIdx.x * 256 + threadIdx.x;    // B_*8*32*8 = 262144
  int col = idx & 7, ch = (idx >> 3) & 31, r = (idx >> 8) & 7, b = idx >> 11;
  float g[4];
#pragma unroll
  for (int gi = 0; gi < 4; ++gi) {
    int oc = gi * CH_ + ch;
    float acc = cxb[oc] + chb[oc];
    for (int k = 0; k < 3; ++k) {
      int cc = col + k - 1;
      if (cc < 0 || cc >= COLS_) continue;
      if (layer0) {
        acc += cxw[oc * 3 + k] * x[((size_t)b * T_ + t) * 64 + r * 8 + cc];
      } else {
        const float* xb  = x + (size_t)b * D_ + r * 256 + cc;
        const float* wxb = cxw + (size_t)(oc * CH_) * 3 + k;
        for (int ic = 0; ic < CH_; ++ic) acc += wxb[ic * 3] * xb[ic * 8];
      }
      const float* hb  = hin + (size_t)b * D_ + r * 256 + cc;
      const float* whb = chw + (size_t)(oc * CH_) * 3 + k;
      for (int ic = 0; ic < CH_; ++ic) acc += whb[ic * 3] * hb[ic * 8];
    }
    g[gi] = acc;
  }
  float cn = sigm(g[1]) * c[idx] + sigm(g[0]) * tanhf(g[3]);
  c[idx] = cn;
  hout[idx] = sigm(g[2]) * tanhf(cn);
}

// ---------------- attention helpers ----------------

// scores[b,s] = enc[b,s,:] . state[b,:]   (one wave per (b,s))
__global__ void inter_scores_kernel(const __bf16* __restrict__ enc, const float* __restrict__ st,
                                    float* __restrict__ sc) {
  int job = blockIdx.x * 8 + (threadIdx.x >> 5);
  int lane = threadIdx.x & 31;
  int b = job >> 7, s = job & (S_ - 1);
  const __bf16* e = enc + ((size_t)b * S_ + s) * D_;
  const float*  v = st + (size_t)b * D_;
  float sum = 0.f;
  for (int j = lane; j < D_; j += 32) sum += (float)e[j] * v[j];
  sum = wave_sum(sum);
  if (lane == 0) sc[b * S_ + s] = sum;
}

// scores[b,i] = rhist[i,b,:] . state[b,:] for i < t  (one wave per (b,i))
__global__ void self_scores_kernel(const __bf16* __restrict__ rh, const float* __restrict__ st,
                                   float* __restrict__ sc, int t) {
  int job = blockIdx.x * 8 + (threadIdx.x >> 5);
  int lane = threadIdx.x & 31;
  int b = job >> 6, i = job & (T_ - 1);
  if (i >= t) return;
  const __bf16* r = rh + ((size_t)i * B_ + b) * D_;
  const float*  v = st + (size_t)b * D_;
  float sum = 0.f;
  for (int j = lane; j < D_; j += 32) sum += (float)r[j] * v[j];
  sum = wave_sum(sum);
  if (lane == 0) sc[b * T_ + i] = sum;
}

// in-place softmax over L entries per row; entries i>=t masked to -1e9 when t>=0
template <int L>
__global__ void softmax_kernel(float* __restrict__ sc, int t) {
  __shared__ float red[L];
  int b = blockIdx.x, i = threadIdx.x;
  float v = (t < 0 || i < t) ? sc[b * L + i] : -1e9f;
  red[i] = v; __syncthreads();
  for (int s = L / 2; s > 0; s >>= 1) { if (i < s) red[i] = fmaxf(red[i], red[i + s]); __syncthreads(); }
  float mx = red[0]; __syncthreads();
  float e = __expf(v - mx);
  red[i] = e; __syncthreads();
  for (int s = L / 2; s > 0; s >>= 1) { if (i < s) red[i] += red[i + s]; __syncthreads(); }
  sc[b * L + i] = e / red[0];
}

// ctx[b,d] = sum_s a[b,s] * enc[b,s,d]
__global__ void inter_ctx_kernel(const __bf16* __restrict__ enc, const float* __restrict__ a,
                                 float* __restrict__ ctx) {
  __shared__ float aw[S_];
  int b = blockIdx.y;
  int d = blockIdx.x * 256 + threadIdx.x;
  if (threadIdx.x < S_) aw[threadIdx.x] = a[b * S_ + threadIdx.x];
  __syncthreads();
  const __bf16* e = enc + (size_t)b * S_ * D_ + d;
  float s = 0.f;
#pragma unroll 4
  for (int j = 0; j < S_; ++j) s += aw[j] * (float)e[(size_t)j * D_];
  ctx[(size_t)b * D_ + d] = s;
}

// ctx[b,d] = sum_{i<t} a[b,i] * phist[i,b,d]   (empty loop -> 0 at t==0, matches (t>0) factor)
__global__ void self_ctx_kernel(const __bf16* __restrict__ ph, const float* __restrict__ a,
                                float* __restrict__ ctx, int t) {
  __shared__ float aw[T_];
  int b = blockIdx.y;
  int d = blockIdx.x * 256 + threadIdx.x;
  if (threadIdx.x < T_) aw[threadIdx.x] = a[b * T_ + threadIdx.x];
  __syncthreads();
  float s = 0.f;
  for (int j = 0; j < t; ++j) s += aw[j] * (float)ph[((size_t)j * B_ + b) * D_ + d];
  ctx[(size_t)b * D_ + d] = s;
}

// cat[b, 0:2D] = bf16([state, ctx])
__global__ void build_cat_kernel(const float* __restrict__ st, const float* __restrict__ ctx,
                                 __bf16* __restrict__ cat) {
  int i = blockIdx.x * 256 + threadIdx.x;      // B_*K2D
  int b = i >> 12, j = i & (K2D - 1);
  float v = (j < D_) ? st[(size_t)b * D_ + j] : ctx[(size_t)b * D_ + (j - D_)];
  cat[i] = (__bf16)v;
}

// ---------------- WMMA GEMM: out = tanh(A(128x4096,bf16) @ W + bias) ----------------
// Wt is (D_ x K2D) N-major bf16. Writes f32 result and a bf16 copy (history slice).
// Block = 8 waves; each wave owns one 16x16 N-tile, all share one 16-row A tile in LDS.
#define LDSK 256
#define LDS_STRIDE (LDSK + 8)

__global__ __launch_bounds__(256)
void gemm_tanh_kernel(const __bf16* __restrict__ A, const __bf16* __restrict__ Wt,
                      const float* __restrict__ bias,
                      float* __restrict__ outf, __bf16* __restrict__ outb) {
  __shared__ __bf16 As[16 * LDS_STRIDE];
  const int wave  = threadIdx.x >> 5;
  const int lane  = threadIdx.x & 31;
  const int mtile = blockIdx.x * 16;
  const int ntile = (blockIdx.y * 8 + wave) * 16;
  const int m  = lane & 15;
  const int hk = lane >> 4;
  const int n  = ntile + (lane & 15);

  v8f acc = {};
  for (int kb = 0; kb < K2D; kb += LDSK) {
    {   // cooperative stage of A tile: 16 rows x 256 K (bf16)
      int row  = threadIdx.x >> 4;
      int colb = (threadIdx.x & 15) << 4;
      const __bf16* src = A + (size_t)(mtile + row) * K2D + kb + colb;
      __bf16* dst = &As[row * LDS_STRIDE + colb];
      *(v8bf*)dst       = *(const v8bf*)src;
      *(v8bf*)(dst + 8) = *(const v8bf*)(src + 8);
    }
    __syncthreads();
#pragma unroll
    for (int kk = 0; kk < LDSK; kk += 32) {
      // A fragment: elem j -> K = kk + hk*16 + j (contiguous)
      v8bf a_lo = *(const v8bf*)&As[m * LDS_STRIDE + kk + hk * 16];
      v8bf a_hi = *(const v8bf*)&As[m * LDS_STRIDE + kk + hk * 16 + 8];
      // B fragment: elem j -> K = kk + hk*16 + j (contiguous in N-major Wt)
      const __bf16* wp = Wt + (size_t)n * K2D + kb + kk + hk * 16;
      v8bf b_lo = *(const v8bf*)wp;
      v8bf b_hi = *(const v8bf*)(wp + 8);
      v16bf av, bv;
#pragma unroll
      for (int j = 0; j < 8; ++j) {
        av[j] = a_lo[j]; av[j + 8] = a_hi[j];
        bv[j] = b_lo[j]; bv[j + 8] = b_hi[j];
      }
      acc = __builtin_amdgcn_wmma_f32_16x16x32_bf16(false, av, false, bv,
                                                    (short)0, acc, false, false);
    }
    __syncthreads();
  }
  float bs = bias[n];
#pragma unroll
  for (int v = 0; v < 8; ++v) {
    int mm = mtile + v + (hk << 3);
    float val = tanhf(acc[v] + bs);
    outf[(size_t)mm * D_ + n] = val;
    outb[(size_t)mm * D_ + n] = (__bf16)val;
  }
}

// ---------------- MLP head ----------------
__global__ void fc_kernel(const float* __restrict__ in, const float* __restrict__ w,
                          const float* __restrict__ bias, float* __restrict__ out,
                          int In, int Out, int ldo, int relu, int n) {
  int i = blockIdx.x * 256 + threadIdx.x;
  if (i >= n) return;
  int b = i / Out, j = i % Out;
  float s = bias[j];
  for (int d = 0; d < In; ++d) s += in[(size_t)b * In + d] * w[(size_t)d * Out + j];
  out[(size_t)b * ldo + j] = relu ? fmaxf(s, 0.f) : s;
}

// ---------------- host ----------------
extern "C" void kernel_launch(void* const* d_in, const int* in_sizes, int n_in,
                              void* d_out, int out_size, void* d_ws, size_t ws_size,
                              hipStream_t stream) {
  (void)in_sizes; (void)n_in; (void)out_size; (void)ws_size;
  const float* x_flat = (const float*)d_in[0];
  const float* enc    = (const float*)d_in[1];
  const float* h0     = (const float*)d_in[2];
  const float* c0     = (const float*)d_in[3];
  const float* cx_w0  = (const float*)d_in[4];
  const float* cx_b0  = (const float*)d_in[5];
  const float* ch_w0  = (const float*)d_in[6];
  const float* ch_b0  = (const float*)d_in[7];
  const float* cx_w1  = (const float*)d_in[8];
  const float* cx_b1  = (const float*)d_in[9];
  const float* ch_w1  = (const float*)d_in[10];
  const float* ch_b1  = (const float*)d_in[11];
  const float* ia_w0  = (const float*)d_in[12];
  const float* ia_b0  = (const float*)d_in[13];
  const float* sa_w0  = (const float*)d_in[14];
  const float* sa_b0  = (const float*)d_in[15];
  const float* ia_w1  = (const float*)d_in[16];
  const float* ia_b1  = (const float*)d_in[17];
  const float* sa_w1  = (const float*)d_in[18];
  const float* sa_b1  = (const float*)d_in[19];
  const float* hw1    = (const float*)d_in[20];
  const float* hb1    = (const float*)d_in[21];
  const float* hw2    = (const float*)d_in[22];
  const float* hb2    = (const float*)d_in[23];
  const float* hw3    = (const float*)d_in[24];
  const float* hb3    = (const float*)d_in[25];
  float* out = (float*)d_out;

  char* cur = (char*)d_ws;
  auto alloc = [&](size_t bytes) -> void* {
    void* r = (void*)cur;
    cur += (bytes + 255) & ~(size_t)255;
    return r;
  };
  const size_t BD = (size_t)B_ * D_;
  __bf16* enc_b  = (__bf16*)alloc((size_t)B_ * S_ * D_ * 2);
  __bf16* wt_ia0 = (__bf16*)alloc((size_t)D_ * K2D * 2);
  __bf16* wt_sa0 = (__bf16*)alloc((size_t)D_ * K2D * 2);
  __bf16* wt_ia1 = (__bf16*)alloc((size_t)D_ * K2D * 2);
  __bf16* wt_sa1 = (__bf16*)alloc((size_t)D_ * K2D * 2);
  float* hAbuf[2] = {(float*)alloc(BD * 4), (float*)alloc(BD * 4)};
  float* hBbuf[2] = {(float*)alloc(BD * 4), (float*)alloc(BD * 4)};
  float* cA  = (float*)alloc(BD * 4);
  float* cB  = (float*)alloc(BD * 4);
  float* hi  = (float*)alloc(BD * 4);
  float* hr0 = (float*)alloc(BD * 4);
  float* hr1 = (float*)alloc(BD * 4);
  float* ctx = (float*)alloc(BD * 4);
  __bf16* cat = (__bf16*)alloc((size_t)B_ * K2D * 2);
  float* scores = (float*)alloc((size_t)B_ * S_ * 4);
  __bf16* p0h = (__bf16*)alloc((size_t)T_ * BD * 2);
  __bf16* r0h = (__bf16*)alloc((size_t)T_ * BD * 2);
  __bf16* p1h = (__bf16*)alloc((size_t)T_ * BD * 2);
  __bf16* r1h = (__bf16*)alloc((size_t)T_ * BD * 2);
  float* z1 = (float*)alloc((size_t)B_ * 200 * 4);
  float* z2 = (float*)alloc((size_t)B_ * 50 * 4);

  // setup: bf16 encoder, transposed bf16 weights, initial states
  cvt_bf16_kernel<<<(B_ * S_ * D_) / 256, 256, 0, stream>>>(enc, enc_b, B_ * S_ * D_);
  cvt_transpose_kernel<<<(D_ * K2D) / 256, 256, 0, stream>>>(ia_w0, wt_ia0);
  cvt_transpose_kernel<<<(D_ * K2D) / 256, 256, 0, stream>>>(sa_w0, wt_sa0);
  cvt_transpose_kernel<<<(D_ * K2D) / 256, 256, 0, stream>>>(ia_w1, wt_ia1);
  cvt_transpose_kernel<<<(D_ * K2D) / 256, 256, 0, stream>>>(sa_w1, wt_sa1);
  init_state_kernel<<<(B_ * D_) / 256, 256, 0, stream>>>(h0, c0, hAbuf[0], cA, hBbuf[0], cB);

  for (int t = 0; t < T_; ++t) {
    int p = t & 1;
    // ---- layer 0 ----
    convlstm_kernel<<<1024, 256, 0, stream>>>(x_flat, 1, t, hAbuf[p], cA, hAbuf[p ^ 1],
                                              cx_w0, cx_b0, ch_w0, ch_b0);
    float* hraw0 = hAbuf[p ^ 1];
    inter_scores_kernel<<<2048, 256, 0, stream>>>(enc_b, hraw0, scores);
    softmax_kernel<128><<<128, 128, 0, stream>>>(scores, -1);
    inter_ctx_kernel<<<dim3(8, 128), 256, 0, stream>>>(enc_b, scores, ctx);
    build_cat_kernel<<<2048, 256, 0, stream>>>(hraw0, ctx, cat);
    gemm_tanh_kernel<<<dim3(8, 16), 256, 0, stream>>>(cat, wt_ia0, ia_b0, hi, p0h + (size_t)t * BD);
    self_scores_kernel<<<1024, 256, 0, stream>>>(r0h, hi, scores, t);
    softmax_kernel<64><<<128, 64, 0, stream>>>(scores, t);
    self_ctx_kernel<<<dim3(8, 128), 256, 0, stream>>>(p0h, scores, ctx, t);
    build_cat_kernel<<<2048, 256, 0, stream>>>(hi, ctx, cat);
    gemm_tanh_kernel<<<dim3(8, 16), 256, 0, stream>>>(cat, wt_sa0, sa_b0, hr0, r0h + (size_t)t * BD);
    // ---- layer 1 ----
    convlstm_kernel<<<1024, 256, 0, stream>>>(hr0, 0, t, hBbuf[p], cB, hBbuf[p ^ 1],
                                              cx_w1, cx_b1, ch_w1, ch_b1);
    float* hraw1 = hBbuf[p ^ 1];
    inter_scores_kernel<<<2048, 256, 0, stream>>>(enc_b, hraw1, scores);
    softmax_kernel<128><<<128, 128, 0, stream>>>(scores, -1);
    inter_ctx_kernel<<<dim3(8, 128), 256, 0, stream>>>(enc_b, scores, ctx);
    build_cat_kernel<<<2048, 256, 0, stream>>>(hraw1, ctx, cat);
    gemm_tanh_kernel<<<dim3(8, 16), 256, 0, stream>>>(cat, wt_ia1, ia_b1, hi, p1h + (size_t)t * BD);
    self_scores_kernel<<<1024, 256, 0, stream>>>(r1h, hi, scores, t);
    softmax_kernel<64><<<128, 64, 0, stream>>>(scores, t);
    self_ctx_kernel<<<dim3(8, 128), 256, 0, stream>>>(p1h, scores, ctx, t);
    build_cat_kernel<<<2048, 256, 0, stream>>>(hi, ctx, cat);
    gemm_tanh_kernel<<<dim3(8, 16), 256, 0, stream>>>(cat, wt_sa1, sa_b1, hr1, r1h + (size_t)t * BD);
    // ---- prediction head ----
    fc_kernel<<<100, 256, 0, stream>>>(hr1, hw1, hb1, z1, D_, 200, 200, 1, B_ * 200);
    fc_kernel<<<25, 256, 0, stream>>>(z1, hw2, hb2, z2, 200, 50, 50, 1, B_ * 50);
    fc_kernel<<<1, 256, 0, stream>>>(z2, hw3, hb3, out + t * 2, 50, 2, T_ * 2, 0, B_ * 2);
  }
}
